// StockPredictor_68736656605721
// MI455X (gfx1250) — matrix operand order repaired and checked by reading, not compiled, not run on hardware
//
#include <hip/hip_runtime.h>

typedef __attribute__((ext_vector_type(16))) _Float16 v16h;
typedef __attribute__((ext_vector_type(8)))  _Float16 v8h;
typedef __attribute__((ext_vector_type(8)))  float    v8f;

constexpr int B  = 256;
constexpr int T  = 2048;
constexpr int H1 = 128;
constexpr int H2 = 64;
constexpr int G1 = 4 * H1;  // 512 gate outputs layer 1
constexpr int G2 = 4 * H2;  // 256 gate outputs layer 2

struct SharedMem {
  _Float16 Whh1[G1][H1];      // 131072 B  (row-major [N][K] == B-matrix layout)
  _Float16 Wih2[G2][H1];      //  65536 B
  _Float16 Whh2[G2][H2];      //  32768 B
  float    b1[G1];            //   2048 B  (b_ih1 + b_hh1)
  float    w1[G1];            //   2048 B  (W_ih1 column; D == 1)
  float    b2[G2];            //   1024 B
  _Float16 h1buf[2][16][H1];  //   8192 B  (double-buffered h1, f16)
  _Float16 h2buf[2][16][H2];  //   4096 B
  float    xbuf[2][16];       //    128 B  (double-buffered x_t per row)
  float    hfin[16][H2];      //   4096 B  (final h2 in f32 for FC head)
};                            // ~251 KB total (< 320 KB per WGP)

// Hardware transcendental tanh (gfx1250 TRANS op). Builtin if the toolchain
// declares it, else inline asm with trailing v_nop to satisfy the TRANS-op
// result-use hazard (1 independent op / V_NOP required after a trans op).
__device__ inline float htanh(float x) {
#if __has_builtin(__builtin_amdgcn_tanhf)
  return __builtin_amdgcn_tanhf(x);
#else
  float r;
  asm volatile("v_tanh_f32 %0, %1\n\tv_nop" : "=v"(r) : "v"(x));
  return r;
#endif
}
// sigmoid(x) = 0.5*tanh(x/2) + 0.5  -> single TRANS op + fma
__device__ inline float sig(float x) { return fmaf(0.5f, htanh(0.5f * x), 0.5f); }

// A-fragment (16xK row-major f16 in LDS), ISA 7.12.2 16-bit A layout:
// lane m=lane&15 is row; hs=lane>>4; VGPR0-3 = K+8*hs..+7, VGPR4-7 = K+16+8*hs..+7
__device__ inline v16h load_a(const _Float16* buf, int K, int kbase, int m, int hs) {
  const _Float16* rp = buf + m * K + kbase + 8 * hs;
  union { v16h v; v8h h[2]; } u;
  u.h[0] = *(const v8h*)(rp);
  u.h[1] = *(const v8h*)(rp + 16);
  return u.v;
}

// B-fragment from one row of row-major [N][K] f16 weights:
// lane n=lane&15 is column N, lane holds K = kbase+8*hs..+7 and kbase+16+8*hs..+7
__device__ inline v16h load_b(const _Float16* Wrow, int kbase, int hs) {
  const _Float16* wp = Wrow + kbase + 8 * hs;
  union { v16h v; v8h h[2]; } u;
  u.h[0] = *(const v8h*)(wp);
  u.h[1] = *(const v8h*)(wp + 16);
  return u.v;
}

__device__ inline v8f wmma_f16(v16h a, v16h b, v8f c) {
  return __builtin_amdgcn_wmma_f32_16x16x32_f16(false, a, false, b, (short)0, c,
                                                false, false);
}

__global__ __launch_bounds__(256) void lstm2_persistent(
    const float* __restrict__ x,
    const float* __restrict__ W_ih1, const float* __restrict__ W_hh1,
    const float* __restrict__ b_ih1, const float* __restrict__ b_hh1,
    const float* __restrict__ W_ih2, const float* __restrict__ W_hh2,
    const float* __restrict__ b_ih2, const float* __restrict__ b_hh2,
    const float* __restrict__ W_fc1, const float* __restrict__ b_fc1,
    const float* __restrict__ W_fc2, const float* __restrict__ b_fc2,
    float* __restrict__ out) {
  __shared__ SharedMem sm;

  const int tid  = threadIdx.x;
  const int wv   = tid >> 5;       // wave id 0..7
  const int lane = tid & 31;
  const int n    = lane & 15;      // N column / A row within tile
  const int hs   = lane >> 4;      // half-wave select
  const int b0   = blockIdx.x * 16;

  // ---- one-time: convert weights f32 -> f16 into LDS ----
  {
    _Float16* d = &sm.Whh1[0][0];
    for (int i = tid; i < G1 * H1; i += 256) d[i] = (_Float16)W_hh1[i];
    d = &sm.Wih2[0][0];
    for (int i = tid; i < G2 * H1; i += 256) d[i] = (_Float16)W_ih2[i];
    d = &sm.Whh2[0][0];
    for (int i = tid; i < G2 * H2; i += 256) d[i] = (_Float16)W_hh2[i];
  }
  for (int i = tid; i < G1; i += 256) {
    sm.b1[i] = b_ih1[i] + b_hh1[i];
    sm.w1[i] = W_ih1[i];            // D == 1: x contribution is outer product
  }
  for (int i = tid; i < G2; i += 256) sm.b2[i] = b_ih2[i] + b_hh2[i];
  { // zero h_{-1} buffers (parity 1), load x_0 (parity 0)
    _Float16* d = &sm.h1buf[1][0][0];
    for (int i = tid; i < 16 * H1; i += 256) d[i] = (_Float16)0.0f;
    d = &sm.h2buf[1][0][0];
    for (int i = tid; i < 16 * H2; i += 256) d[i] = (_Float16)0.0f;
    if (tid < 16) sm.xbuf[0][tid] = x[(size_t)(b0 + tid) * T];
  }
  __syncthreads();

  // per-lane constants: gate bias / x-weight for owned columns
  float bi1[4], wi1[4];
#pragma unroll
  for (int g = 0; g < 4; ++g) {
    const int c = H1 * g + 16 * wv + n;
    bi1[g] = sm.b1[c];
    wi1[g] = sm.w1[c];
  }
  float bi2[4] = {0.f, 0.f, 0.f, 0.f};
  if (wv < 4) {
#pragma unroll
    for (int g = 0; g < 4; ++g) bi2[g] = sm.b2[H2 * g + 16 * wv + n];
  }

  v8f c1 = {};     // cell state L1, cols [16*wv, 16*wv+16), rows r+8*hs
  v8f c2 = {};     // cell state L2 (waves 0..3 only)
  v8f h2reg = {};  // final h2 in f32 (waves 0..3)

  for (int t = 0; t < T; ++t) {
    const int p = t & 1;

    // ---------- layer 1: gates1[16x512] = h1_{t-1} @ Whh1^T + x_t*w1 + b1 ----
    v16h a1[4];
    const _Float16* h1p = &sm.h1buf[p ^ 1][0][0];
#pragma unroll
    for (int k = 0; k < 4; ++k) a1[k] = load_a(h1p, H1, 32 * k, n, hs);

    float xr[8];
#pragma unroll
    for (int r = 0; r < 8; ++r) xr[r] = sm.xbuf[p][r + 8 * hs];

    v8f acc[4];
#pragma unroll
    for (int g = 0; g < 4; ++g) {
#pragma unroll
      for (int r = 0; r < 8; ++r) acc[g][r] = bi1[g] + xr[r] * wi1[g];
    }
#pragma unroll
    for (int g = 0; g < 4; ++g) {
      const _Float16* Wrow = &sm.Whh1[H1 * g + 16 * wv + n][0];
#pragma unroll
      for (int k = 0; k < 4; ++k)
        acc[g] = wmma_f16(a1[k], load_b(Wrow, 32 * k, hs), acc[g]);
    }

    // ---------- LSTM cell 1 (registers; hw tanh transcendentals) ----------
    v8f h1new;
#pragma unroll
    for (int r = 0; r < 8; ++r) {
      const float iv = sig(acc[0][r]);
      const float fv = sig(acc[1][r]);
      const float gv = htanh(acc[2][r]);
      const float ov = sig(acc[3][r]);
      const float cc = fv * c1[r] + iv * gv;
      c1[r] = cc;
      h1new[r] = ov * htanh(cc);
    }
#pragma unroll
    for (int r = 0; r < 8; ++r)
      sm.h1buf[p][r + 8 * hs][16 * wv + n] = (_Float16)h1new[r];

    // prefetch x_{t+1} into the other parity (consumed after next barrier)
    if (wv == 7 && lane < 16) {
      const int tt = (t + 1 < T) ? (t + 1) : t;
      sm.xbuf[p ^ 1][lane] = x[(size_t)(b0 + lane) * T + tt];
    }
    __syncthreads();  // single barrier per step (everything double-buffered)

    // ---------- layer 2 (waves 0..3): gates2 = h1_t@Wih2^T + h2_{t-1}@Whh2^T + b2
    if (wv < 4) {
      v16h a1n[4], a2[2];
      const _Float16* h1n = &sm.h1buf[p][0][0];
#pragma unroll
      for (int k = 0; k < 4; ++k) a1n[k] = load_a(h1n, H1, 32 * k, n, hs);
      const _Float16* h2p = &sm.h2buf[p ^ 1][0][0];
#pragma unroll
      for (int k = 0; k < 2; ++k) a2[k] = load_a(h2p, H2, 32 * k, n, hs);

      v8f acc2[4];
#pragma unroll
      for (int g = 0; g < 4; ++g) {
#pragma unroll
        for (int r = 0; r < 8; ++r) acc2[g][r] = bi2[g];
      }
#pragma unroll
      for (int g = 0; g < 4; ++g) {
        const int nrow = H2 * g + 16 * wv + n;
        const _Float16* Wr1 = &sm.Wih2[nrow][0];
#pragma unroll
        for (int k = 0; k < 4; ++k)
          acc2[g] = wmma_f16(a1n[k], load_b(Wr1, 32 * k, hs), acc2[g]);
        const _Float16* Wr2 = &sm.Whh2[nrow][0];
#pragma unroll
        for (int k = 0; k < 2; ++k)
          acc2[g] = wmma_f16(a2[k], load_b(Wr2, 32 * k, hs), acc2[g]);
      }
#pragma unroll
      for (int r = 0; r < 8; ++r) {
        const float iv = sig(acc2[0][r]);
        const float fv = sig(acc2[1][r]);
        const float gv = htanh(acc2[2][r]);
        const float ov = sig(acc2[3][r]);
        const float cc = fv * c2[r] + iv * gv;
        c2[r] = cc;
        const float hh = ov * htanh(cc);
        h2reg[r] = hh;
        sm.h2buf[p][r + 8 * hs][16 * wv + n] = (_Float16)hh;
      }
    }
  }

  // ---------- FC head on final h2 (f32 precision kept from registers) ----------
  if (wv < 4) {
#pragma unroll
    for (int r = 0; r < 8; ++r) sm.hfin[r + 8 * hs][16 * wv + n] = h2reg[r];
  }
  __syncthreads();
  if (tid < 16) {
    float o = b_fc2[0];
    for (int k = 0; k < 25; ++k) {
      float s = b_fc1[k];
      for (int j = 0; j < H2; ++j) s += W_fc1[k * H2 + j] * sm.hfin[tid][j];
      o += W_fc2[k] * s;
    }
    out[b0 + tid] = o;
  }
}

extern "C" void kernel_launch(void* const* d_in, const int* in_sizes, int n_in,
                              void* d_out, int out_size, void* d_ws, size_t ws_size,
                              hipStream_t stream) {
  (void)in_sizes; (void)n_in; (void)out_size; (void)d_ws; (void)ws_size;
  const float* x     = (const float*)d_in[0];
  const float* W_ih1 = (const float*)d_in[1];
  const float* W_hh1 = (const float*)d_in[2];
  const float* b_ih1 = (const float*)d_in[3];
  const float* b_hh1 = (const float*)d_in[4];
  const float* W_ih2 = (const float*)d_in[5];
  const float* W_hh2 = (const float*)d_in[6];
  const float* b_ih2 = (const float*)d_in[7];
  const float* b_hh2 = (const float*)d_in[8];
  const float* W_fc1 = (const float*)d_in[9];
  const float* b_fc1 = (const float*)d_in[10];
  const float* W_fc2 = (const float*)d_in[11];
  const float* b_fc2 = (const float*)d_in[12];
  float* out = (float*)d_out;

  lstm2_persistent<<<B / 16, 256, 0, stream>>>(
      x, W_ih1, W_hh1, b_ih1, b_hh1, W_ih2, W_hh2, b_ih2, b_hh2,
      W_fc1, b_fc1, W_fc2, b_fc2, out);
}
